// WeaveNet_25941602468191
// MI455X (gfx1250) — matrix operand order repaired and checked by reading, not compiled, not run on hardware
//
#include <hip/hip_runtime.h>

typedef __attribute__((ext_vector_type(16))) __bf16 v16bf;
typedef __attribute__((ext_vector_type(8)))  __bf16 v8bf;
typedef __attribute__((ext_vector_type(2)))  __bf16 v2bf;
typedef __attribute__((ext_vector_type(8)))  float  v8f;

#define N_NODES 50000
#define N_EDGES 800000
#define ND 128
#define ED 32
#define H  64
#define L_LAYERS 4

#if defined(__gfx1250__) && __has_builtin(__builtin_amdgcn_tensor_load_to_lds)
#define HAVE_TDM 1
#else
#define HAVE_TDM 0
#endif
#if defined(__gfx1250__) && __has_builtin(__builtin_amdgcn_global_load_async_to_lds_b128)
#define HAVE_ASYNC 1
#else
#define HAVE_ASYNC 0
#endif

// ---------- helpers ----------
__device__ __forceinline__ unsigned pack2(float a, float b) {
  v2bf t;
  t[0] = (__bf16)a;   // native RTNE convert -> v_cvt(_pk)_bf16_f32
  t[1] = (__bf16)b;
  return __builtin_bit_cast(unsigned, t);
}
__device__ __forceinline__ unsigned short f2bf(float x) {
  __bf16 t = (__bf16)x;
  return __builtin_bit_cast(unsigned short, t);
}
__device__ __forceinline__ v8f wmma_bf16(v16bf a, v16bf b, v8f c) {
  return __builtin_amdgcn_wmma_f32_16x16x32_bf16(false, a, false, b, (short)0, c,
                                                 false, false);
}
// A operand: weight^T row-major [M][Ktot] bf16 in LDS.
// lane: M = mt*16 + lane%16; K = kc*32 + 16*half + 0..15  (contiguous 32B)
__device__ __forceinline__ v16bf load_a(const unsigned short* w, int m, int Ktot,
                                        int kc, int half) {
  const __bf16* p = (const __bf16*)w + m * Ktot + kc * 32 + half * 16;
  return *(const v16bf*)p;
}
// B operand: activation row-major [16][Ktot] bf16 in LDS.
// lane: N = lane%16; K = kc*32 + 16*half + 0..15  (contiguous 32B)
__device__ __forceinline__ v16bf load_b(const unsigned short* act, int n, int Ktot,
                                        int kc, int half) {
  const __bf16* p = (const __bf16*)act + n * Ktot + kc * 32 + half * 16;
  return *(const v16bf*)p;
}
__device__ __forceinline__ void wait_lds() {
  asm volatile("s_wait_dscnt 0x0" ::: "memory");
}

#if HAVE_TDM
typedef unsigned int u32x4 __attribute__((ext_vector_type(4)));
typedef int          i32x8 __attribute__((ext_vector_type(8)));
typedef int          i32x4 __attribute__((ext_vector_type(4)));
typedef __attribute__((address_space(3))) const void* lds_cvptr;

__device__ __forceinline__ unsigned lds_off(const void* p) {
  return (unsigned)(unsigned long long)(lds_cvptr)p;
}
// 1-row 2D tile descriptor: copy n dwords global -> LDS via Tensor Data Mover.
__device__ __forceinline__ void tdm_load_dwords(const void* g, unsigned lds_byte_off,
                                                unsigned n) {
  unsigned long long ga = (unsigned long long)(size_t)g;
  u32x4 g0;
  g0[0] = 1u;                                                // count=1, user mode
  g0[1] = lds_byte_off;                                      // lds_addr (bytes)
  g0[2] = (unsigned)ga;                                      // global_addr[31:0]
  g0[3] = (unsigned)((ga >> 32) & 0x01ffffffu) | (2u << 30); // ga[56:32] | type=2
  i32x8 g1;
  g1[0] = 2 << 16;                                  // data_size=4B, no multicast
  g1[1] = (int)((n & 0xffffu) << 16);               // tensor_dim0[15:0]
  g1[2] = (int)((n >> 16) & 0xffffu) | (1 << 16);   // tensor_dim0[31:16] | tensor_dim1=1
  g1[3] = (int)(n << 16);                           // tile_dim0 = n (n <= 65535)
  g1[4] = 1;                                        // tile_dim1=1, tile_dim2=0
  g1[5] = (int)n;                                   // tensor_dim0_stride lo
  g1[6] = 0;
  g1[7] = 0;
  i32x4 z4 = {0, 0, 0, 0};
  i32x8 z8 = {0, 0, 0, 0, 0, 0, 0, 0};
  __builtin_amdgcn_tensor_load_to_lds(g0, g1, z4, z4, z8, 0);
}
#endif

#if HAVE_ASYNC
typedef int v4i_g __attribute__((vector_size(16)));
typedef __attribute__((address_space(1))) v4i_g* gv4i_ptr;
typedef __attribute__((address_space(3))) v4i_g* lv4i_ptr;
__device__ __forceinline__ void wait_async() {
  asm volatile("s_wait_asynccnt 0x0" ::: "memory");
}
#endif

// ---------- prep: transpose fp32 [b][K][M] -> bf16 [b][M][K] ----------
__global__ void k_prep(const float* __restrict__ in, unsigned short* __restrict__ out,
                       int K, int M, int total) {
  for (int idx = blockIdx.x * blockDim.x + threadIdx.x; idx < total;
       idx += gridDim.x * blockDim.x) {
    int b = idx / (K * M);
    int r = idx - b * K * M;
    int k = r / M;
    int m = r - k * M;
    out[(size_t)b * K * M + (size_t)m * K + k] = f2bf(in[idx]);
  }
}

__global__ void k_zero(float* __restrict__ p, int n) {
  for (int i = blockIdx.x * blockDim.x + threadIdx.x; i < n;
       i += gridDim.x * blockDim.x) p[i] = 0.f;
}

// ---------- node embedding: h = x @ node_W + b  (x:[N,128] -> h:[N,64] fp32) ----------
__global__ void __launch_bounds__(128) k_embed_nodes(
    const float* __restrict__ x, const unsigned short* __restrict__ WT,
    const float* __restrict__ bias, float* __restrict__ h) {
  __shared__ __align__(32) unsigned short sWT[H * ND];
  __shared__ float sB[H];
  __shared__ __align__(32) unsigned short sAct[4][16 * ND];

#if HAVE_TDM
  if ((threadIdx.x >> 5) == 0) {
    tdm_load_dwords(WT, lds_off(sWT), H * ND / 2);
    __builtin_amdgcn_s_wait_tensorcnt(0);
  }
#else
  for (int i = threadIdx.x; i < H * ND; i += blockDim.x) sWT[i] = WT[i];
#endif
  for (int i = threadIdx.x; i < H; i += blockDim.x) sB[i] = bias[i];
  __syncthreads();

  const int wave = threadIdx.x >> 5, lane = threadIdx.x & 31;
  const int half = lane >> 4, l16 = lane & 15;
  const int n0 = (blockIdx.x * 4 + wave) * 16;
  if (n0 >= N_NODES) return;

  {  // stage 16 node rows as bf16 [16][128]; lane covers feats [64*half, +64)
    const int node = n0 + l16;
    const float4* s4 = (const float4*)(x + (size_t)node * ND + 64 * half);
    uint2* q = (uint2*)(sAct[wave] + l16 * ND + 64 * half);
#pragma unroll
    for (int j = 0; j < 16; ++j) {
      float4 f = s4[j];
      q[j] = make_uint2(pack2(f.x, f.y), pack2(f.z, f.w));
    }
  }
  wait_lds();

  v8f zero = {0.f, 0.f, 0.f, 0.f, 0.f, 0.f, 0.f, 0.f};
  v8f acc[4] = {zero, zero, zero, zero};
#pragma unroll
  for (int kc = 0; kc < 4; ++kc) {
    v16bf b = load_b(sAct[wave], l16, ND, kc, half);
#pragma unroll
    for (int mt = 0; mt < 4; ++mt) {
      v16bf a = load_a(sWT, mt * 16 + l16, ND, kc, half);
      acc[mt] = wmma_bf16(a, b, acc[mt]);
    }
  }
  const int node = n0 + l16;
#pragma unroll
  for (int mt = 0; mt < 4; ++mt) {
    const int fb = mt * 16 + half * 8;
    float* o = h + (size_t)node * H + fb;
#pragma unroll
    for (int v = 0; v < 8; ++v) o[v] = acc[mt][v] + sB[fb + v];
  }
}

// ---------- edge embedding: ea = edge_attr @ edge_W + b  -> bf16 [E][64] ----------
__global__ void __launch_bounds__(128) k_embed_edges(
    const float* __restrict__ ein, const unsigned short* __restrict__ WT,
    const float* __restrict__ bias, unsigned short* __restrict__ ea) {
  __shared__ __align__(32) unsigned short sWT[H * ED];
  __shared__ float sB[H];
  __shared__ __align__(32) unsigned short sAct[4][16 * ED];

#if HAVE_TDM
  if ((threadIdx.x >> 5) == 0) {
    tdm_load_dwords(WT, lds_off(sWT), H * ED / 2);
    __builtin_amdgcn_s_wait_tensorcnt(0);
  }
#else
  for (int i = threadIdx.x; i < H * ED; i += blockDim.x) sWT[i] = WT[i];
#endif
  for (int i = threadIdx.x; i < H; i += blockDim.x) sB[i] = bias[i];
  __syncthreads();

  const int wave = threadIdx.x >> 5, lane = threadIdx.x & 31;
  const int half = lane >> 4, l16 = lane & 15;
  const int e0 = (blockIdx.x * 4 + wave) * 16;

  {  // stage 16 edge_attr rows bf16 [16][32]; lane covers feats [16*half, +16)
    const float4* s4 = (const float4*)(ein + (size_t)(e0 + l16) * ED + 16 * half);
    uint2* q = (uint2*)(sAct[wave] + l16 * ED + 16 * half);
#pragma unroll
    for (int j = 0; j < 4; ++j) {
      float4 f = s4[j];
      q[j] = make_uint2(pack2(f.x, f.y), pack2(f.z, f.w));
    }
  }
  wait_lds();

  v8f zero = {0.f, 0.f, 0.f, 0.f, 0.f, 0.f, 0.f, 0.f};
  v8f acc[4] = {zero, zero, zero, zero};
  v16bf b = load_b(sAct[wave], l16, ED, 0, half);
#pragma unroll
  for (int mt = 0; mt < 4; ++mt) {
    v16bf a = load_a(sWT, mt * 16 + l16, ED, 0, half);
    acc[mt] = wmma_bf16(a, b, acc[mt]);
  }
  const int e = e0 + l16;
#pragma unroll
  for (int mt = 0; mt < 4; ++mt) {
    const int fb = mt * 16 + half * 8;
    uint4 o;
    o.x = pack2(acc[mt][0] + sB[fb + 0], acc[mt][1] + sB[fb + 1]);
    o.y = pack2(acc[mt][2] + sB[fb + 2], acc[mt][3] + sB[fb + 3]);
    o.z = pack2(acc[mt][4] + sB[fb + 4], acc[mt][5] + sB[fb + 5]);
    o.w = pack2(acc[mt][6] + sB[fb + 6], acc[mt][7] + sB[fb + 7]);
    *(uint4*)(ea + (size_t)e * H + fb) = o;
  }
}

// ---------- message-passing layer ----------
__global__ void __launch_bounds__(128) k_layer(
    const float* __restrict__ h_in, float* __restrict__ h_out,
    const unsigned short* __restrict__ ea,
    const int* __restrict__ srcI, const int* __restrict__ dstI,
    const unsigned short* __restrict__ W1T, const float* __restrict__ b1,
    const unsigned short* __restrict__ W2T, const float* __restrict__ b2) {
  __shared__ __align__(32) unsigned short sW1T[H * 2 * H];  // 64x128
  __shared__ __align__(32) unsigned short sW2T[H * H];      // 64x64
  __shared__ float sB1[H], sB2[H];
  __shared__ __align__(32) unsigned short sMsg[4][16 * 128];
  __shared__ __align__(32) unsigned short sM1[4][16 * 64];

#if HAVE_TDM
  if ((threadIdx.x >> 5) == 0) {
    tdm_load_dwords(W1T, lds_off(sW1T), H * 2 * H / 2);  // 4096 dwords
    tdm_load_dwords(W2T, lds_off(sW2T), H * H / 2);      // 2048 dwords
    __builtin_amdgcn_s_wait_tensorcnt(0);
  }
#else
  for (int i = threadIdx.x; i < H * 2 * H; i += blockDim.x) sW1T[i] = W1T[i];
  for (int i = threadIdx.x; i < H * H; i += blockDim.x) sW2T[i] = W2T[i];
#endif
  for (int i = threadIdx.x; i < H; i += blockDim.x) { sB1[i] = b1[i]; sB2[i] = b2[i]; }
  __syncthreads();

  const int wave = threadIdx.x >> 5, lane = threadIdx.x & 31;
  const int half = lane >> 4, l16 = lane & 15;
  const int e0 = (blockIdx.x * 4 + wave) * 16;
  const int e = e0 + l16;
  const int s = srcI[e], d = dstI[e];

  {  // build msg row [l16]: cols 0..63 = bf16(h[s]+h[d]), cols 64..127 = ea
    const int fb = 32 * half;
    const float4* ps = (const float4*)(h_in + (size_t)s * H + fb);
    const float4* pd = (const float4*)(h_in + (size_t)d * H + fb);
    uint2* q = (uint2*)(sMsg[wave] + l16 * 128 + fb);
#pragma unroll
    for (int j = 0; j < 8; ++j) {
      float4 a = ps[j], c = pd[j];
      q[j] = make_uint2(pack2(a.x + c.x, a.y + c.y), pack2(a.z + c.z, a.w + c.w));
    }
#if HAVE_ASYNC
    gv4i_ptr gp = (gv4i_ptr)(ea + (size_t)e * H + fb);
    lv4i_ptr lp = (lv4i_ptr)(sMsg[wave] + l16 * 128 + 64 + fb);
    __builtin_amdgcn_global_load_async_to_lds_b128(gp, lp, 0, 0);
    __builtin_amdgcn_global_load_async_to_lds_b128(gp, lp, 16, 0);
    __builtin_amdgcn_global_load_async_to_lds_b128(gp, lp, 32, 0);
    __builtin_amdgcn_global_load_async_to_lds_b128(gp, lp, 48, 0);
#else
    const uint4* pe = (const uint4*)(ea + (size_t)e * H + fb);
    uint4* qe = (uint4*)(sMsg[wave] + l16 * 128 + 64 + fb);
#pragma unroll
    for (int j = 0; j < 4; ++j) qe[j] = pe[j];
#endif
  }
#if HAVE_ASYNC
  wait_async();
#endif
  wait_lds();

  // GEMM1: m1^T = W1^T [64x128] x msg^T [128x16]
  v8f zero = {0.f, 0.f, 0.f, 0.f, 0.f, 0.f, 0.f, 0.f};
  v8f acc[4] = {zero, zero, zero, zero};
#pragma unroll
  for (int kc = 0; kc < 4; ++kc) {
    v16bf b = load_b(sMsg[wave], l16, 128, kc, half);
#pragma unroll
    for (int mt = 0; mt < 4; ++mt) {
      v16bf a = load_a(sW1T, mt * 16 + l16, 128, kc, half);
      acc[mt] = wmma_bf16(a, b, acc[mt]);
    }
  }
  // relu(+b1) -> bf16 m1 [16 edges][64 feats] in LDS
#pragma unroll
  for (int mt = 0; mt < 4; ++mt) {
    const int fb = mt * 16 + half * 8;
    float r[8];
#pragma unroll
    for (int v = 0; v < 8; ++v) {
      float t = acc[mt][v] + sB1[fb + v];
      r[v] = t > 0.f ? t : 0.f;
    }
    uint4 o;
    o.x = pack2(r[0], r[1]); o.y = pack2(r[2], r[3]);
    o.z = pack2(r[4], r[5]); o.w = pack2(r[6], r[7]);
    *(uint4*)(sM1[wave] + l16 * H + fb) = o;
  }
  wait_lds();

  // GEMM2: m2^T = W2^T [64x64] x m1^T [64x16]
  v8f acc2[4] = {zero, zero, zero, zero};
#pragma unroll
  for (int kc = 0; kc < 2; ++kc) {
    v16bf b = load_b(sM1[wave], l16, H, kc, half);
#pragma unroll
    for (int mt = 0; mt < 4; ++mt) {
      v16bf a = load_a(sW2T, mt * 16 + l16, H, kc, half);
      acc2[mt] = wmma_bf16(a, b, acc2[mt]);
    }
  }
  // scatter-add at destination node (hardware f32 atomics)
  float* hb = h_out + (size_t)d * H;
#pragma unroll
  for (int mt = 0; mt < 4; ++mt) {
    const int fb = mt * 16 + half * 8;
#pragma unroll
    for (int v = 0; v < 8; ++v)
      unsafeAtomicAdd(hb + fb + v, acc2[mt][v] + sB2[fb + v]);
  }
}

// ---------- final scorer ----------
__global__ void __launch_bounds__(128) k_final(
    const float* __restrict__ h,
    const int* __restrict__ srcI, const int* __restrict__ dstI,
    const unsigned short* __restrict__ W1T, const float* __restrict__ b1,
    const float* __restrict__ w2, const float* __restrict__ b2,
    float* __restrict__ out) {
  __shared__ __align__(32) unsigned short sW1T[H * 2 * H];
  __shared__ float sB1[H], sW2[H];
  __shared__ float sB2s;
  __shared__ __align__(32) unsigned short sMsg[4][16 * 128];

#if HAVE_TDM
  if ((threadIdx.x >> 5) == 0) {
    tdm_load_dwords(W1T, lds_off(sW1T), H * 2 * H / 2);
    __builtin_amdgcn_s_wait_tensorcnt(0);
  }
#else
  for (int i = threadIdx.x; i < H * 2 * H; i += blockDim.x) sW1T[i] = W1T[i];
#endif
  for (int i = threadIdx.x; i < H; i += blockDim.x) { sB1[i] = b1[i]; sW2[i] = w2[i]; }
  if (threadIdx.x == 0) sB2s = b2[0];
  __syncthreads();

  const int wave = threadIdx.x >> 5, lane = threadIdx.x & 31;
  const int half = lane >> 4, l16 = lane & 15;
  const int e0 = (blockIdx.x * 4 + wave) * 16;
  const int e = e0 + l16;

  {  // lanes 0-15: h[src] -> cols 0..63 ; lanes 16-31: h[dst] -> cols 64..127
    const int idx = half ? dstI[e] : srcI[e];
    const float4* s4 = (const float4*)(h + (size_t)idx * H);
    uint2* q = (uint2*)(sMsg[wave] + l16 * 128 + half * 64);
#pragma unroll
    for (int j = 0; j < 16; ++j) {
      float4 f = s4[j];
      q[j] = make_uint2(pack2(f.x, f.y), pack2(f.z, f.w));
    }
  }
  wait_lds();

  v8f zero = {0.f, 0.f, 0.f, 0.f, 0.f, 0.f, 0.f, 0.f};
  v8f acc[4] = {zero, zero, zero, zero};
#pragma unroll
  for (int kc = 0; kc < 4; ++kc) {
    v16bf b = load_b(sMsg[wave], l16, 128, kc, half);
#pragma unroll
    for (int mt = 0; mt < 4; ++mt) {
      v16bf a = load_a(sW1T, mt * 16 + l16, 128, kc, half);
      acc[mt] = wmma_bf16(a, b, acc[mt]);
    }
  }
  float p = 0.f;
#pragma unroll
  for (int mt = 0; mt < 4; ++mt) {
    const int fb = mt * 16 + half * 8;
#pragma unroll
    for (int v = 0; v < 8; ++v) {
      float t = acc[mt][v] + sB1[fb + v];
      t = t > 0.f ? t : 0.f;
      p += t * sW2[fb + v];
    }
  }
  p += __shfl_xor(p, 16, 32);  // combine the two half-lane partials per edge
  if (half == 0) out[e] = p + sB2s;
}

// ---------- host ----------
extern "C" void kernel_launch(void* const* d_in, const int* in_sizes, int n_in,
                              void* d_out, int out_size, void* d_ws, size_t ws_size,
                              hipStream_t stream) {
  const float* x         = (const float*)d_in[0];
  const float* edge_attr = (const float*)d_in[1];
  const int*   ei        = (const int*)d_in[2];
  const float* node_W    = (const float*)d_in[3];
  const float* node_b    = (const float*)d_in[4];
  const float* edge_W    = (const float*)d_in[5];
  const float* edge_b    = (const float*)d_in[6];
  const float* lW1       = (const float*)d_in[7];
  const float* lb1       = (const float*)d_in[8];
  const float* lW2       = (const float*)d_in[9];
  const float* lb2       = (const float*)d_in[10];
  const float* fW1       = (const float*)d_in[11];
  const float* fb1       = (const float*)d_in[12];
  const float* fW2       = (const float*)d_in[13];
  const float* fb2       = (const float*)d_in[14];
  float* out = (float*)d_out;

  char* ws = (char*)d_ws;
  size_t off = 0;
  auto alloc = [&](size_t bytes) -> char* {
    char* p = ws + off;
    off = (off + bytes + 255) & ~(size_t)255;
    return p;
  };
  float* hA = (float*)alloc((size_t)N_NODES * H * 4);
  float* hB = (float*)alloc((size_t)N_NODES * H * 4);
  unsigned short* ea     = (unsigned short*)alloc((size_t)N_EDGES * H * 2);
  unsigned short* nodeWT = (unsigned short*)alloc((size_t)ND * H * 2);
  unsigned short* edgeWT = (unsigned short*)alloc((size_t)ED * H * 2);
  unsigned short* W1T    = (unsigned short*)alloc((size_t)L_LAYERS * 2 * H * H * 2);
  unsigned short* W2T    = (unsigned short*)alloc((size_t)L_LAYERS * H * H * 2);
  unsigned short* finW1T = (unsigned short*)alloc((size_t)2 * H * H * 2);

  // weight prep: fp32 [K][M] -> bf16 [M][K]
  k_prep<<<32, 256, 0, stream>>>(node_W, nodeWT, ND, H, ND * H);
  k_prep<<<8, 256, 0, stream>>>(edge_W, edgeWT, ED, H, ED * H);
  k_prep<<<128, 256, 0, stream>>>(lW1, W1T, 2 * H, H, L_LAYERS * 2 * H * H);
  k_prep<<<64, 256, 0, stream>>>(lW2, W2T, H, H, L_LAYERS * H * H);
  k_prep<<<32, 256, 0, stream>>>(fW1, finW1T, 2 * H, H, 2 * H * H);

  const int nodeTiles = N_NODES / 16;       // 3125
  const int edgeBlocks = N_EDGES / 16 / 4;  // 12500
  k_embed_nodes<<<(nodeTiles + 3) / 4, 128, 0, stream>>>(x, nodeWT, node_b, hA);
  k_embed_edges<<<edgeBlocks, 128, 0, stream>>>(edge_attr, edgeWT, edge_b, ea);

  const int* srcI = ei;
  const int* dstI = ei + N_EDGES;
  float* hin = hA;
  float* hout = hB;
  for (int l = 0; l < L_LAYERS; ++l) {
    k_zero<<<2048, 256, 0, stream>>>(hout, N_NODES * H);
    k_layer<<<edgeBlocks, 128, 0, stream>>>(
        hin, hout, ea, srcI, dstI,
        W1T + (size_t)l * 2 * H * H, lb1 + (size_t)l * H,
        W2T + (size_t)l * H * H, lb2 + (size_t)l * H);
    float* t = hin; hin = hout; hout = t;
  }
  k_final<<<edgeBlocks, 128, 0, stream>>>(hin, srcI, dstI, finW1T, fb1, fW2, fb2, out);
}
